// BIMPM_76347338654261
// MI455X (gfx1250) — matrix operand order repaired and checked by reading, not compiled, not run on hardware
//
#include <hip/hip_runtime.h>
#include <hip/hip_bf16.h>
#include <math.h>

typedef __attribute__((ext_vector_type(16))) _Float16 v16h;
typedef __attribute__((ext_vector_type(8)))  _Float16 h8;
typedef __attribute__((ext_vector_type(8)))  float    v8f;

#define EPSV 1e-8f
#define Bq 32
#define Sq 64
#define Eq 300
#define EP 320    // padded K for embeddings (multiple of 32)
#define Hq 256
#define Lq 20
#define Gq 1024
#define MPW 160   // 8*L

__device__ __forceinline__ float sigf(float x){ return 1.0f/(1.0f+expf(-x)); }

// ---- WMMA fragment loaders from pre-packed f16, no masking, aligned vector loads ----
// A: 16x32 (MxK). lane: m = mbase+(lane&15); halves: K = khi + i (i<8), khi+16+(i-8) (i>=8)
// where khi = kbase + 8*(lane>>4). Two aligned 16B loads.
__device__ __forceinline__ v16h frag_a16(const _Float16* A, int mbase, int kbase,
                                         int lda, int lane){
  const _Float16* p = A + (size_t)(mbase + (lane & 15)) * lda
                        + kbase + ((lane >> 4) & 1) * 8;
  h8 lo = *(const h8*)(p);
  h8 hi = *(const h8*)(p + 16);
  return __builtin_shufflevector(lo, hi, 0,1,2,3,4,5,6,7,8,9,10,11,12,13,14,15);
}
// B: 32x16 (KxN) with B[k][n] = W[n][k] (W row-major NxK f16).
// lane: n = nbase+(lane&15); halves: K = khi + i (i<8), khi+16+(i-8) (i>=8), khi = kbase + 8*(lane>>4).
__device__ __forceinline__ v16h frag_b16(const _Float16* __restrict__ W, int nbase,
                                         int kbase, int ldw, int lane){
  const _Float16* p = W + (size_t)(nbase + (lane & 15)) * ldw
                        + kbase + ((lane >> 4) & 1) * 8;
  h8 lo = *(const h8*)(p);
  h8 hi = *(const h8*)(p + 16);
  return __builtin_shufflevector(lo, hi, 0,1,2,3,4,5,6,7,8,9,10,11,12,13,14,15);
}

// ---- packing kernels (one-time f32 -> padded f16) ----
__global__ void k_embed_f16(const int* __restrict__ q, const float* __restrict__ emb,
                            _Float16* __restrict__ out){
  int i = blockIdx.x*blockDim.x + threadIdx.x;
  if(i >= Bq*Sq*EP) return;
  int e = i % EP; int r = i / EP;
  out[i] = (e < Eq) ? (_Float16)emb[(size_t)q[r]*Eq + e] : (_Float16)0.0f;
}

__global__ void k_pack16(const float* __restrict__ src, _Float16* __restrict__ dst,
                         int R, int K, int Kp){
  int i = blockIdx.x*blockDim.x + threadIdx.x;
  if(i >= R*Kp) return;
  int k = i % Kp; int r = i / Kp;
  dst[i] = (k < K) ? (_Float16)src[(size_t)r*K + k] : (_Float16)0.0f;
}

// ---- out[m][n] = sum_k X[m][k]*W[n][k] + b1[n] + b2[n]  (WMMA f16->f32) ----
// block = 128 thr (4 waves); wave computes 16x64 tile; grid (M/16, N/256). Kp % 32 == 0.
__global__ void k_gemm16(const _Float16* __restrict__ X, const _Float16* __restrict__ W,
                         const float* __restrict__ b1v, const float* __restrict__ b2v,
                         float* __restrict__ out, int M, int N, int Kp){
  int lane = threadIdx.x & 31;
  int wave = threadIdx.x >> 5;
  int mbase = blockIdx.x * 16;
  int nbase = blockIdx.y * 256 + wave * 64;
  v8f acc[4];
  #pragma unroll
  for(int j=0;j<4;++j){
    #pragma unroll
    for(int v=0;v<8;++v) acc[j][v] = 0.0f;
  }
  for(int kb=0; kb<Kp; kb+=32){
    v16h a = frag_a16(X, mbase, kb, Kp, lane);
    #pragma unroll
    for(int j=0;j<4;++j){
      v16h b = frag_b16(W, nbase + j*16, kb, Kp, lane);
      acc[j] = __builtin_amdgcn_wmma_f32_16x16x32_f16(false, a, false, b,
                                                      (short)0, acc[j], false, false);
    }
  }
  #pragma unroll
  for(int j=0;j<4;++j){
    int n = nbase + j*16 + (lane & 15);
    float bb = b1v[n] + b2v[n];
    #pragma unroll
    for(int v=0;v<8;++v){
      int m = mbase + v + 8*(lane >> 4);
      out[(size_t)m*N + n] = acc[j][v] + bb;
    }
  }
}

// ---- LSTM scan: one 1024-thread workgroup per sequence/direction; state in LDS ----
struct ScanPtrs { const float* xg; const _Float16* wh; float* hs; int rev; };
struct ScanArgs { ScanPtrs c[4]; };

#define HP2 264   // padded LDS stride (halves) for h16: 528B rows, 16B aligned
#define HPC 260   // padded LDS stride (floats) for c
#define GP  1028  // padded LDS stride (floats) for gates
// layout: h16 (32*HP2 halves = 16896B) | c (32*HPC f32) | g (32*GP f32)
#define C_OFF   (32*HP2/2)           // float offset of c region
#define G_OFF   (C_OFF + 32*HPC)     // float offset of g region
#define SCAN_LDS_BYTES ((G_OFF + 32*GP)*4)

__global__ void k_lstm_scan(ScanArgs A){
  extern __shared__ float smem[];
  _Float16* h16  = (_Float16*)smem;      // 32 rows * HP2 halves
  float*    c_buf = smem + C_OFF;        // 32 rows * HPC
  float*    g_buf = smem + G_OFF;        // 32 rows * GP
  const ScanPtrs sp = A.c[blockIdx.x];
  int tid = threadIdx.x, lane = tid & 31, wave = tid >> 5;
  for(int r=0;r<8;++r){
    int idx = tid + r*1024; int b = idx >> 8; int u = idx & 255;
    h16[b*HP2+u] = (_Float16)0.0f; c_buf[b*HPC+u] = 0.0f;
  }
  __syncthreads();
  int mtile = wave >> 4;     // 0..1 -> batch rows 0-15 / 16-31
  int nbase = (wave & 15) * 64;
  for(int t=0; t<Sq; ++t){
    int tt = sp.rev ? (Sq-1-t) : t;
    // g = h @ wh^T  (32x1024, K=256)
    v8f acc[4];
    #pragma unroll
    for(int j=0;j<4;++j){
      #pragma unroll
      for(int v=0;v<8;++v) acc[j][v] = 0.0f;
    }
    #pragma unroll
    for(int kb=0; kb<Hq; kb+=32){
      v16h a = frag_a16(h16, mtile*16, kb, HP2, lane);
      #pragma unroll
      for(int j=0;j<4;++j){
        v16h b = frag_b16(sp.wh, nbase + j*16, kb, Hq, lane);
        acc[j] = __builtin_amdgcn_wmma_f32_16x16x32_f16(false, a, false, b,
                                                        (short)0, acc[j], false, false);
      }
    }
    #pragma unroll
    for(int j=0;j<4;++j){
      int n = nbase + j*16 + (lane & 15);
      #pragma unroll
      for(int v=0;v<8;++v){
        int m = mtile*16 + v + 8*(lane >> 4);
        g_buf[m*GP + n] = acc[j][v] + sp.xg[((size_t)m*Sq + tt)*Gq + n];
      }
    }
    __syncthreads();
    // gates + state update (32*256 elems, 8 per thread)
    for(int r=0;r<8;++r){
      int idx = tid + r*1024; int b = idx >> 8; int u = idx & 255;
      float gi = g_buf[b*GP + u];
      float gf = g_buf[b*GP + 256 + u];
      float gg = g_buf[b*GP + 512 + u];
      float go = g_buf[b*GP + 768 + u];
      float cn = sigf(gf)*c_buf[b*HPC+u] + sigf(gi)*tanhf(gg);
      float hn = sigf(go)*tanhf(cn);
      c_buf[b*HPC+u] = cn;
      h16[b*HP2+u] = (_Float16)hn;
      sp.hs[((size_t)b*Sq + tt)*Hq + u] = hn;
    }
    __syncthreads();
  }
}

// ---- cosine matching / attention (elementwise VALU kernels) ----
__global__ void k_rownorm(const float* __restrict__ v, float* __restrict__ rn){
  int i = blockIdx.x*blockDim.x + threadIdx.x;
  if(i >= Bq*Sq) return;
  float s = 0.0f;
  const float* x = v + (size_t)i*Hq;
  for(int h=0;h<Hq;++h) s += x[h]*x[h];
  rn[i] = sqrtf(s);
}

__global__ void k_mpfull(const float* __restrict__ v1, const float* __restrict__ v2base,
                         int v2_sstride, const float* __restrict__ w,
                         float* __restrict__ mv, int colOff){
  int i = blockIdx.x*blockDim.x + threadIdx.x;
  if(i >= Bq*Sq*Lq) return;
  int l = i % Lq; int s = (i/Lq) % Sq; int b = i/(Lq*Sq);
  const float* x1 = v1 + ((size_t)b*Sq + s)*Hq;
  const float* x2 = v2base + (size_t)b*(Sq*Hq) + (size_t)s*v2_sstride;
  const float* wl = w + (size_t)l*Hq;
  float num=0.f, na=0.f, nb=0.f;
  for(int h=0;h<Hq;++h){
    float aa = x1[h]*wl[h];
    float bb = x2[h]*wl[h];
    num += aa*bb; na += aa*aa; nb += bb*bb;
  }
  float den = fmaxf(sqrtf(na)*sqrtf(nb), EPSV);
  mv[((size_t)b*Sq + s)*MPW + colOff + l] = num/den;
}

__global__ void k_pairnorm(const float* __restrict__ v, const float* __restrict__ w,
                           float* __restrict__ pn){
  int i = blockIdx.x*blockDim.x + threadIdx.x;
  if(i >= Bq*Lq*Sq) return;
  int s = i % Sq; int l = (i/Sq) % Lq; int b = i/(Sq*Lq);
  const float* x = v + ((size_t)b*Sq + s)*Hq;
  const float* wl = w + (size_t)l*Hq;
  float sm = 0.f;
  for(int h=0;h<Hq;++h){ float a = x[h]*wl[h]; sm += a*a; }
  pn[i] = sqrtf(sm);
}

__global__ void k_pairmax(const float* __restrict__ v1, const float* __restrict__ v2,
                          const float* __restrict__ w, const float* __restrict__ nA,
                          const float* __restrict__ nB, float* __restrict__ mv, int colOff){
  int i = blockIdx.x*blockDim.x + threadIdx.x;
  if(i >= Bq*Sq*Lq) return;
  int l = i % Lq; int s = (i/Lq) % Sq; int b = i/(Lq*Sq);
  const float* x1 = v1 + ((size_t)b*Sq + s)*Hq;
  const float* wl = w + (size_t)l*Hq;
  float na = nA[((size_t)b*Lq + l)*Sq + s];
  float best = -3.4e38f;
  for(int t=0;t<Sq;++t){
    const float* x2 = v2 + ((size_t)b*Sq + t)*Hq;
    float num = 0.f;
    for(int h=0;h<Hq;++h) num += wl[h]*wl[h]*x1[h]*x2[h];
    float d = na * nB[((size_t)b*Lq + l)*Sq + t];
    d = (d > EPSV) ? d : EPSV;
    best = fmaxf(best, num/d);
  }
  mv[((size_t)b*Sq + s)*MPW + colOff + l] = best;
}

__global__ void k_att(const float* __restrict__ v1, const float* __restrict__ v2,
                      const float* __restrict__ rn1, const float* __restrict__ rn2,
                      float* __restrict__ att){
  int i = blockIdx.x*blockDim.x + threadIdx.x;
  if(i >= Bq*Sq*Sq) return;
  int t = i % Sq; int s = (i/Sq) % Sq; int b = i/(Sq*Sq);
  const float* x1 = v1 + ((size_t)b*Sq + s)*Hq;
  const float* x2 = v2 + ((size_t)b*Sq + t)*Hq;
  float num = 0.f;
  for(int h=0;h<Hq;++h) num += x1[h]*x2[h];
  float d = rn1[b*Sq+s]*rn2[b*Sq+t];
  d = (d > EPSV) ? d : EPSV;
  att[i] = num/d;
}

__global__ void k_attstats(const float* __restrict__ att, const float* __restrict__ V,
                           float* __restrict__ outMean, float* __restrict__ outMax, int overS){
  int i = blockIdx.x*blockDim.x + threadIdx.x;
  if(i >= Bq*Sq*Hq) return;
  int hh = i % Hq; int so = (i/Hq) % Sq; int b = i/(Hq*Sq);
  float sumv=0.f, sumw=0.f, mx=-3.4e38f;
  for(int k=0;k<Sq;++k){
    float a = overS ? att[((size_t)b*Sq + k)*Sq + so] : att[((size_t)b*Sq + so)*Sq + k];
    float vv = V[((size_t)b*Sq + k)*Hq + hh] * a;
    sumv += vv; sumw += a; mx = fmaxf(mx, vv);
  }
  float d = (sumw > EPSV) ? sumw : EPSV;
  outMean[((size_t)b*Sq + so)*Hq + hh] = sumv/d;
  outMax [((size_t)b*Sq + so)*Hq + hh] = mx;
}

// ---- final aggregation -> logits ----
__global__ void k_concat(const float* __restrict__ h0, const float* __restrict__ h1,
                         const float* __restrict__ h2, const float* __restrict__ h3,
                         float* __restrict__ xcat){
  int i = blockIdx.x*blockDim.x + threadIdx.x;
  if(i >= Bq*4*Hq) return;
  int j = i % (4*Hq); int b = i / (4*Hq);
  int part = j >> 8; int u = j & 255;
  float v;
  if(part == 0)      v = h0[((size_t)b*Sq + (Sq-1))*Hq + u];
  else if(part == 1) v = h1[((size_t)b*Sq + 0)*Hq + u];
  else if(part == 2) v = h2[((size_t)b*Sq + (Sq-1))*Hq + u];
  else               v = h3[((size_t)b*Sq + 0)*Hq + u];
  xcat[i] = v;
}

__global__ void k_fc1(const float* __restrict__ xcat, const float* __restrict__ w1,
                      const float* __restrict__ b1, float* __restrict__ x1v){
  int i = blockIdx.x*blockDim.x + threadIdx.x;
  if(i >= Bq*512) return;
  int j = i % 512; int b = i / 512;
  const float* x = xcat + (size_t)b*1024;
  const float* wr = w1 + (size_t)j*1024;
  float s = b1[j];
  for(int k=0;k<1024;++k) s += x[k]*wr[k];
  x1v[i] = tanhf(s);
}

__global__ void k_fc2(const float* __restrict__ x1v, const float* __restrict__ w2,
                      const float* __restrict__ b2, float* __restrict__ out){
  int b = threadIdx.x;
  if(b >= Bq) return;
  const float* x = x1v + (size_t)b*512;
  float l0 = b2[0], l1 = b2[1];
  for(int k=0;k<512;++k){ l0 += x[k]*w2[k]; l1 += x[k]*w2[512+k]; }
  out[b*2+0] = l0;
  out[b*2+1] = l1;
  out[Bq*2 + b] = (l1 > l0) ? 1.0f : 0.0f;
}

extern "C" void kernel_launch(void* const* d_in, const int* in_sizes, int n_in,
                              void* d_out, int out_size, void* d_ws, size_t ws_size,
                              hipStream_t stream) {
  (void)in_sizes; (void)n_in; (void)out_size; (void)ws_size;
  const int*   q1    = (const int*)d_in[0];
  const int*   q2    = (const int*)d_in[1];
  const float* emb   = (const float*)d_in[2];
  const float* cwi_f = (const float*)d_in[3];
  const float* cwh_f = (const float*)d_in[4];
  const float* cbi_f = (const float*)d_in[5];
  const float* cbh_f = (const float*)d_in[6];
  const float* cwi_b = (const float*)d_in[7];
  const float* cwh_b = (const float*)d_in[8];
  const float* cbi_b = (const float*)d_in[9];
  const float* cbh_b = (const float*)d_in[10];
  const float* mp_w  = (const float*)d_in[11];
  const float* awi_f = (const float*)d_in[12];
  const float* awh_f = (const float*)d_in[13];
  const float* abi_f = (const float*)d_in[14];
  const float* abh_f = (const float*)d_in[15];
  const float* awi_b = (const float*)d_in[16];
  const float* awh_b = (const float*)d_in[17];
  const float* abi_b = (const float*)d_in[18];
  const float* abh_b = (const float*)d_in[19];
  const float* w1    = (const float*)d_in[20];
  const float* b1    = (const float*)d_in[21];
  const float* w2    = (const float*)d_in[22];
  const float* b2    = (const float*)d_in[23];

  char* base = (char*)d_ws;
  size_t off = 0;
  auto allocf = [&](size_t n)->float* {
    off = (off + 255) & ~(size_t)255;
    float* p = (float*)(base + off); off += n*4; return p;
  };
  auto alloch = [&](size_t n)->_Float16* {
    off = (off + 255) & ~(size_t)255;
    _Float16* p = (_Float16*)(base + off); off += n*2; return p;
  };

  // f32 buffers
  float* xg0 = allocf((size_t)Bq*Sq*Gq);   // reused for aggregation phase
  float* xg1 = allocf((size_t)Bq*Sq*Gq);
  float* xg2 = allocf((size_t)Bq*Sq*Gq);
  float* xg3 = allocf((size_t)Bq*Sq*Gq);
  float* conPf = allocf((size_t)Bq*Sq*Hq);
  float* conPb = allocf((size_t)Bq*Sq*Hq);
  float* conHf = allocf((size_t)Bq*Sq*Hq);
  float* conHb = allocf((size_t)Bq*Sq*Hq);
  float* mvp = allocf((size_t)Bq*Sq*MPW);
  float* mvh = allocf((size_t)Bq*Sq*MPW);
  float* attF = allocf((size_t)Bq*Sq*Sq);
  float* attB = allocf((size_t)Bq*Sq*Sq);
  float* rnPf = allocf(Bq*Sq); float* rnPb = allocf(Bq*Sq);
  float* rnHf = allocf(Bq*Sq); float* rnHb = allocf(Bq*Sq);
  float* pnPf = allocf((size_t)Bq*Lq*Sq); float* pnHf = allocf((size_t)Bq*Lq*Sq);
  float* pnPb = allocf((size_t)Bq*Lq*Sq); float* pnHb = allocf((size_t)Bq*Lq*Sq);
  float* meanHf = allocf((size_t)Bq*Sq*Hq); float* maxHf = allocf((size_t)Bq*Sq*Hq);
  float* meanPf = allocf((size_t)Bq*Sq*Hq); float* maxPf = allocf((size_t)Bq*Sq*Hq);
  float* meanHb = allocf((size_t)Bq*Sq*Hq); float* maxHb = allocf((size_t)Bq*Sq*Hq);
  float* meanPb = allocf((size_t)Bq*Sq*Hq); float* maxPb = allocf((size_t)Bq*Sq*Hq);
  float* hs0 = allocf((size_t)Bq*Sq*Hq);
  float* hs1 = allocf((size_t)Bq*Sq*Hq);
  float* hs2 = allocf((size_t)Bq*Sq*Hq);
  float* hs3 = allocf((size_t)Bq*Sq*Hq);
  float* xcat = allocf((size_t)Bq*4*Hq);
  float* x1v  = allocf((size_t)Bq*512);

  // f16 packed buffers
  _Float16* pX     = alloch((size_t)Bq*Sq*EP);
  _Float16* hX     = alloch((size_t)Bq*Sq*EP);
  _Float16* cwiF16 = alloch((size_t)Gq*EP);
  _Float16* cwiB16 = alloch((size_t)Gq*EP);
  _Float16* cwhF16 = alloch((size_t)Gq*Hq);
  _Float16* cwhB16 = alloch((size_t)Gq*Hq);
  _Float16* awiF16 = alloch((size_t)Gq*MPW);
  _Float16* awiB16 = alloch((size_t)Gq*MPW);
  _Float16* awhF16 = alloch((size_t)Gq*Hq);
  _Float16* awhB16 = alloch((size_t)Gq*Hq);
  _Float16* mvp16  = alloch((size_t)Bq*Sq*MPW);
  _Float16* mvh16  = alloch((size_t)Bq*Sq*MPW);

  const float* wl0 = mp_w;
  const float* wl1 = mp_w + 1*Lq*Hq;
  const float* wl2 = mp_w + 2*Lq*Hq;
  const float* wl3 = mp_w + 3*Lq*Hq;
  const float* wl4 = mp_w + 4*Lq*Hq;
  const float* wl5 = mp_w + 5*Lq*Hq;
  const float* wl6 = mp_w + 6*Lq*Hq;
  const float* wl7 = mp_w + 7*Lq*Hq;

  // 1) embed + pack all GEMM operands to padded f16
  int embTot = Bq*Sq*EP;
  k_embed_f16<<<(embTot+255)/256, 256, 0, stream>>>(q1, emb, pX);
  k_embed_f16<<<(embTot+255)/256, 256, 0, stream>>>(q2, emb, hX);
  k_pack16<<<(Gq*EP +255)/256, 256, 0, stream>>>(cwi_f, cwiF16, Gq, Eq, EP);
  k_pack16<<<(Gq*EP +255)/256, 256, 0, stream>>>(cwi_b, cwiB16, Gq, Eq, EP);
  k_pack16<<<(Gq*Hq +255)/256, 256, 0, stream>>>(cwh_f, cwhF16, Gq, Hq, Hq);
  k_pack16<<<(Gq*Hq +255)/256, 256, 0, stream>>>(cwh_b, cwhB16, Gq, Hq, Hq);
  k_pack16<<<(Gq*MPW+255)/256, 256, 0, stream>>>(awi_f, awiF16, Gq, MPW, MPW);
  k_pack16<<<(Gq*MPW+255)/256, 256, 0, stream>>>(awi_b, awiB16, Gq, MPW, MPW);
  k_pack16<<<(Gq*Hq +255)/256, 256, 0, stream>>>(awh_f, awhF16, Gq, Hq, Hq);
  k_pack16<<<(Gq*Hq +255)/256, 256, 0, stream>>>(awh_b, awhB16, Gq, Hq, Hq);

  // 2) context input projections: xg = x@wi^T + bi + bh  (M=2048,N=1024,Kp=320)
  dim3 gg(128, 4);
  k_gemm16<<<gg,128,0,stream>>>(pX, cwiF16, cbi_f, cbh_f, xg0, Bq*Sq, Gq, EP);
  k_gemm16<<<gg,128,0,stream>>>(pX, cwiB16, cbi_b, cbh_b, xg1, Bq*Sq, Gq, EP);
  k_gemm16<<<gg,128,0,stream>>>(hX, cwiF16, cbi_f, cbh_f, xg2, Bq*Sq, Gq, EP);
  k_gemm16<<<gg,128,0,stream>>>(hX, cwiB16, cbi_b, cbh_b, xg3, Bq*Sq, Gq, EP);

  // 3) 4 context LSTM scans, one WGP-resident workgroup each
  ScanArgs sa;
  sa.c[0] = { xg0, cwhF16, conPf, 0 };
  sa.c[1] = { xg1, cwhB16, conPb, 1 };
  sa.c[2] = { xg2, cwhF16, conHf, 0 };
  sa.c[3] = { xg3, cwhB16, conHb, 1 };
  k_lstm_scan<<<4, 1024, SCAN_LDS_BYTES, stream>>>(sa);

  // 4) matching
  k_rownorm<<<8,256,0,stream>>>(conPf, rnPf);
  k_rownorm<<<8,256,0,stream>>>(conPb, rnPb);
  k_rownorm<<<8,256,0,stream>>>(conHf, rnHf);
  k_rownorm<<<8,256,0,stream>>>(conHb, rnHb);
  k_att<<<512,256,0,stream>>>(conPf, conHf, rnPf, rnHf, attF);
  k_att<<<512,256,0,stream>>>(conPb, conHb, rnPb, rnHb, attB);
  k_attstats<<<2048,256,0,stream>>>(attF, conHf, meanHf, maxHf, 0);
  k_attstats<<<2048,256,0,stream>>>(attF, conPf, meanPf, maxPf, 1);
  k_attstats<<<2048,256,0,stream>>>(attB, conHb, meanHb, maxHb, 0);
  k_attstats<<<2048,256,0,stream>>>(attB, conPb, meanPb, maxPb, 1);

  k_mpfull<<<160,256,0,stream>>>(conPf, conHf + (Sq-1)*Hq, 0, wl0, mvp, 0);
  k_mpfull<<<160,256,0,stream>>>(conPb, conHb,             0, wl1, mvp, 80);
  k_mpfull<<<160,256,0,stream>>>(conHf, conPf + (Sq-1)*Hq, 0, wl0, mvh, 0);
  k_mpfull<<<160,256,0,stream>>>(conHb, conPb,             0, wl1, mvh, 80);

  k_pairnorm<<<160,256,0,stream>>>(conPf, wl2, pnPf);
  k_pairnorm<<<160,256,0,stream>>>(conHf, wl2, pnHf);
  k_pairnorm<<<160,256,0,stream>>>(conPb, wl3, pnPb);
  k_pairnorm<<<160,256,0,stream>>>(conHb, wl3, pnHb);
  k_pairmax<<<160,256,0,stream>>>(conPf, conHf, wl2, pnPf, pnHf, mvp, 20);
  k_pairmax<<<160,256,0,stream>>>(conHf, conPf, wl2, pnHf, pnPf, mvh, 20);
  k_pairmax<<<160,256,0,stream>>>(conPb, conHb, wl3, pnPb, pnHb, mvp, 100);
  k_pairmax<<<160,256,0,stream>>>(conHb, conPb, wl3, pnHb, pnPb, mvh, 100);

  k_mpfull<<<160,256,0,stream>>>(conPf, meanHf, Hq, wl4, mvp, 40);
  k_mpfull<<<160,256,0,stream>>>(conPb, meanHb, Hq, wl5, mvp, 120);
  k_mpfull<<<160,256,0,stream>>>(conHf, meanPf, Hq, wl4, mvh, 40);
  k_mpfull<<<160,256,0,stream>>>(conHb, meanPb, Hq, wl5, mvh, 120);
  k_mpfull<<<160,256,0,stream>>>(conPf, maxHf, Hq, wl6, mvp, 60);
  k_mpfull<<<160,256,0,stream>>>(conPb, maxHb, Hq, wl7, mvp, 140);
  k_mpfull<<<160,256,0,stream>>>(conHf, maxPf, Hq, wl6, mvh, 60);
  k_mpfull<<<160,256,0,stream>>>(conHb, maxPb, Hq, wl7, mvh, 140);

  // 5) aggregation input projections (Kp=160), reuse xg buffers
  int mvTot = Bq*Sq*MPW;
  k_pack16<<<(mvTot+255)/256, 256, 0, stream>>>(mvp, mvp16, Bq*Sq, MPW, MPW);
  k_pack16<<<(mvTot+255)/256, 256, 0, stream>>>(mvh, mvh16, Bq*Sq, MPW, MPW);
  k_gemm16<<<gg,128,0,stream>>>(mvp16, awiF16, abi_f, abh_f, xg0, Bq*Sq, Gq, MPW);
  k_gemm16<<<gg,128,0,stream>>>(mvp16, awiB16, abi_b, abh_b, xg1, Bq*Sq, Gq, MPW);
  k_gemm16<<<gg,128,0,stream>>>(mvh16, awiF16, abi_f, abh_f, xg2, Bq*Sq, Gq, MPW);
  k_gemm16<<<gg,128,0,stream>>>(mvh16, awiB16, abi_b, abh_b, xg3, Bq*Sq, Gq, MPW);

  // 6) 4 aggregation LSTM scans
  ScanArgs sb;
  sb.c[0] = { xg0, awhF16, hs0, 0 };
  sb.c[1] = { xg1, awhB16, hs1, 1 };
  sb.c[2] = { xg2, awhF16, hs2, 0 };
  sb.c[3] = { xg3, awhB16, hs3, 1 };
  k_lstm_scan<<<4, 1024, SCAN_LDS_BYTES, stream>>>(sb);

  // 7) head
  k_concat<<<128,256,0,stream>>>(hs0, hs1, hs2, hs3, xcat);
  k_fc1<<<64,256,0,stream>>>(xcat, w1, b1, x1v);
  k_fc2<<<1,32,0,stream>>>(x1v, w2, b2, (float*)d_out);
}
